// StructuredCodebook_59322088292905
// MI455X (gfx1250) — matrix operand order repaired and checked by reading, compiled, not run on hardware
//
#include <hip/hip_runtime.h>

typedef __attribute__((ext_vector_type(16))) __bf16 v16bf;
typedef __attribute__((ext_vector_type(8)))  float  v8f;

#define R_TOTAL 65536   // B*N = 8*8192 rows
#define DDIM    256     // reduction dim
#define KG      1024    // codebook entries
#define TOPK    8
#define NEGV    (-10000.0f)

union FragU { uint4 u[2]; v16bf v; };

// --- pre-pass: codes f32 -> bf16 into workspace (512 KB, stays hot in L2) ---
__global__ __launch_bounds__(256)
void convert_codes_kernel(const float* __restrict__ codes, __bf16* __restrict__ out) {
    int i = (blockIdx.x * 256 + threadIdx.x) * 4;
    float4 f = *reinterpret_cast<const float4*>(codes + i);
    union { __bf16 h[4]; uint2 u; } p;
    p.h[0] = (__bf16)f.x; p.h[1] = (__bf16)f.y;
    p.h[2] = (__bf16)f.z; p.h[3] = (__bf16)f.w;
    *reinterpret_cast<uint2*>(out + i) = p.u;
}

// --- main: 32 rows x 1024 cols per block; 8 waves x 8 col-tiles x 2 row-tiles.
//     Each B fragment feeds TWO v_wmma_f32_16x16x32_bf16 (halves L2 read/FLOP).
//     Scheduler handles load/WMMA overlap; explicit double-buffering caused spills. ---
__global__ __launch_bounds__(256)
void codebook_topk_kernel(const float* __restrict__ q,
                          const __bf16* __restrict__ codesb,
                          const int* __restrict__ mask,
                          float* __restrict__ out_idx,
                          float* __restrict__ out_w,
                          float* __restrict__ out_sim) {
    __shared__ float smem[16 * KG];                    // 64 KB: sim half-tile / q stage overlay
    __bf16* qlds = reinterpret_cast<__bf16*>(smem);    // first 16 KB during GEMM phase

    const int tid  = threadIdx.x;
    const int lane = tid & 31;
    const int wave = tid >> 5;
    const int row0 = blockIdx.x * 32;

    // Phase 1: stage q tile (32x256) as bf16 in LDS, coalesced f32 reads
    #pragma unroll
    for (int it = 0; it < 32; ++it) {
        int i = it * 256 + tid;
        qlds[i] = (__bf16)q[(size_t)row0 * DDIM + i];
    }
    __syncthreads();

    // Phase 2: GEMM. A frag layout: lane<16 -> row=lane%16, K {0..7,16..23};
    //                               lane>=16 ->            K {8..15,24..31}
    v8f acc[2][8];
    #pragma unroll
    for (int r = 0; r < 2; ++r)
        #pragma unroll
        for (int t = 0; t < 8; ++t)
            #pragma unroll
            for (int v = 0; v < 8; ++v) acc[r][t][v] = 0.0f;

    const int arow  = lane & 15;
    const int akoff = (lane < 16) ? 0 : 8;
    const int bkoff = (lane < 16) ? 0 : 8;

    for (int k0 = 0; k0 < DDIM; k0 += 32) {
        FragU a0, a1;
        {
            const __bf16* ap0 = &qlds[(arow)      * DDIM + k0 + akoff];
            const __bf16* ap1 = &qlds[(16 + arow) * DDIM + k0 + akoff];
            a0.u[0] = *reinterpret_cast<const uint4*>(ap0);
            a0.u[1] = *reinterpret_cast<const uint4*>(ap0 + 16);
            a1.u[0] = *reinterpret_cast<const uint4*>(ap1);
            a1.u[1] = *reinterpret_cast<const uint4*>(ap1 + 16);
        }
        #pragma unroll
        for (int t = 0; t < 8; ++t) {
            int n = wave * 128 + t * 16 + (lane & 15);
            FragU b;
            const __bf16* bp = codesb + (size_t)n * DDIM + k0 + bkoff;
            b.u[0] = *reinterpret_cast<const uint4*>(bp);
            b.u[1] = *reinterpret_cast<const uint4*>(bp + 16);
            acc[0][t] = __builtin_amdgcn_wmma_f32_16x16x32_bf16(
                            false, a0.v, false, b.v, (short)0, acc[0][t], false, false);
            acc[1][t] = __builtin_amdgcn_wmma_f32_16x16x32_bf16(
                            false, a1.v, false, b.v, (short)0, acc[1][t], false, false);
        }
    }
    __syncthreads();   // all waves done reading qlds before sim overwrites it

    // Phases 3-5, per 16-row half through the same 64 KB LDS buffer
    #pragma unroll
    for (int half = 0; half < 2; ++half) {
        // Phase 3: accumulators -> LDS sim half-tile.
        // C layout: VGPR v = row M=v (lanes<16) / M=v+8 (lanes>=16), N = lane%16
        #pragma unroll
        for (int t = 0; t < 8; ++t) {
            int n = wave * 128 + t * 16 + (lane & 15);
            int mbase = (lane < 16) ? 0 : 8;
            #pragma unroll
            for (int v = 0; v < 8; ++v)
                smem[(mbase + v) * KG + n] = acc[half][t][v];
        }
        __syncthreads();

        const int hrow0 = row0 + half * 16;

        // Phase 4: coalesced b128 copy of 16x1024 sim half-tile to global (raw, unmasked)
        {
            const uint4* src = reinterpret_cast<const uint4*>(smem);
            uint4* dst = reinterpret_cast<uint4*>(out_sim + (size_t)hrow0 * KG);
            #pragma unroll
            for (int it = 0; it < 16; ++it) {
                int i = it * 256 + tid;
                dst[i] = src[i];
            }
        }

        // Phase 5: top-8 + softmax. Wave w handles local rows 2w, 2w+1.
        for (int rr = 0; rr < 2; ++rr) {
            int rloc = wave * 2 + rr;
            int grow = hrow0 + rloc;
            int mval = mask[grow];

            float vals[32];
            #pragma unroll
            for (int j = 0; j < 32; ++j)
                vals[j] = smem[rloc * KG + j * 32 + lane];   // candidate index = j*32 + lane
            if (mval == 0) {
                #pragma unroll
                for (int j = 0; j < 32; ++j) vals[j] = NEGV; // ties -> idx 0..7, weights 0
            }

            float myv = 0.0f; int myi = 0;
            float topmax = NEGV;
            #pragma unroll
            for (int t = 0; t < TOPK; ++t) {
                // local argmax (strict > keeps lowest index on ties)
                float bv = vals[0]; int bi = lane;
                #pragma unroll
                for (int j = 1; j < 32; ++j) {
                    int id = j * 32 + lane;
                    if (vals[j] > bv) { bv = vals[j]; bi = id; }
                }
                // wave32 butterfly argmax, tie-break to lower index (lax.top_k stability)
                #pragma unroll
                for (int off = 16; off >= 1; off >>= 1) {
                    float ov = __shfl_xor(bv, off, 32);
                    int   oi = __shfl_xor(bi, off, 32);
                    if (ov > bv || (ov == bv && oi < bi)) { bv = ov; bi = oi; }
                }
                if (lane == t) { myv = bv; myi = bi; }
                if (t == 0) topmax = bv;
                // knock out the winner (register-resident, fully unrolled select)
                if ((bi & 31) == lane) {
                    int jj = bi >> 5;
                    #pragma unroll
                    for (int j = 0; j < 32; ++j)
                        if (j == jj) vals[j] = -3.0e38f;
                }
            }

            // softmax over the 8 extracted values (tau = 0.5 -> scale by 2)
            float e = (lane < TOPK) ? __expf((myv - topmax) * 2.0f) : 0.0f;
            float s = e;
            #pragma unroll
            for (int off = 16; off >= 1; off >>= 1) s += __shfl_xor(s, off, 32);
            if (lane < TOPK) {
                float w = (mval != 0) ? (e / s) : 0.0f;
                out_idx[(size_t)grow * TOPK + lane] = (float)myi;
                out_w  [(size_t)grow * TOPK + lane] = w;
            }
        }
        __syncthreads();   // all top-k reads of this half done before next half overwrites LDS
    }
}

extern "C" void kernel_launch(void* const* d_in, const int* in_sizes, int n_in,
                              void* d_out, int out_size, void* d_ws, size_t ws_size,
                              hipStream_t stream) {
    const float* q     = (const float*)d_in[0];
    const float* codes = (const float*)d_in[1];
    const int*   mask  = (const int*)d_in[2];
    // d_in[3] = top_k scalar (fixed 8)

    float* out     = (float*)d_out;
    float* out_idx = out;                                  // [R, 8]
    float* out_w   = out + (size_t)R_TOTAL * TOPK;         // [R, 8]
    float* out_sim = out + (size_t)R_TOTAL * TOPK * 2;     // [R, 1024]

    __bf16* codesb = (__bf16*)d_ws;                        // 512 KB bf16 codes

    convert_codes_kernel<<<(KG * DDIM) / (256 * 4), 256, 0, stream>>>(codes, codesb);
    codebook_topk_kernel<<<R_TOTAL / 32, 256, 0, stream>>>(q, codesb, mask,
                                                           out_idx, out_w, out_sim);
}
